// Quant4Linear_18373870092919
// MI455X (gfx1250) — matrix operand order repaired and checked by reading, not compiled, not run on hardware
//
#include <hip/hip_runtime.h>
#include <stdint.h>

typedef __attribute__((ext_vector_type(16))) _Float16 v16h;
typedef __attribute__((ext_vector_type(8)))  float    v8f;
typedef __attribute__((ext_vector_type(8)))  uint32_t v8u;

#define IN_F   8192
#define OUT_F  28672
#define KCH    256          // IN_F / 32 (K per WMMA)
#define TOKENS 16
#define MN     (TOKENS * OUT_F)

// Permutation: WMMA-K position -> k offset within a 32-wide chunk.
// Chosen so the nibble unpack in the hot loop needs no interleave:
// each packed word (8 nibbles, k = 8g..8g+7) expands as evens-then-odds.
__device__ __forceinline__ int pi_map(int K) {
  int g = K >> 3, j = K & 7;
  return (g << 3) + ((j < 4) ? (j << 1) : (((j - 4) << 1) | 1));
}

// ---- Pre-pass 1: pack x (f32 [16,8192]) into WMMA A fragments (f16),
// K-permuted by pi_map so it matches the B unpack order.
__global__ void q4l_packA(const float* __restrict__ x, _Float16* __restrict__ aws) {
  int tid = blockIdx.x * blockDim.x + threadIdx.x;  // 8192 threads: (chunk, lane)
  int c = tid >> 5;
  int L = tid & 31;
  int m = L & 15;
  _Float16* dst = aws + ((size_t)(c * 32 + L) << 4);
#pragma unroll
  for (int h = 0; h < 16; ++h) {
    int Kpos = (L < 16) ? h : h + 16;
    int k = (c << 5) + pi_map(Kpos);
    dst[h] = (_Float16)x[m * IN_F + k];
  }
}

// ---- Pre-pass 2: per-row sum of f16-rounded x (so the +1024 weight offset
// cancels exactly against the WMMA accumulation).
__global__ void q4l_rowsum(const float* __restrict__ x, float* __restrict__ sums) {
  int r    = threadIdx.x >> 5;   // 16 waves, one row each
  int lane = threadIdx.x & 31;
  float p = 0.f;
  for (int j = lane; j < IN_F; j += 32)
    p += (float)(_Float16)x[r * IN_F + j];
#pragma unroll
  for (int off = 16; off > 0; off >>= 1)
    p += __shfl_xor(p, off, 32);
  if (lane == 0) sums[r] = p;
}

// Expand one packed word (8 nibbles) into 4 dwords of f16 halves,
// each half = 0x6400 | n  ==  (f16)(1024 + n).  Order: (n0,n2)(n4,n6)(n1,n3)(n5,n7).
struct U4 { uint32_t a, b, c, d; };
__device__ __forceinline__ U4 q4l_unpack(uint32_t w) {
  const uint32_t K64 = 0x64646464u;
  uint32_t e = w & 0x0F0F0F0Fu;          // bytes: n0 n2 n4 n6
  uint32_t o = (w >> 4) & 0x0F0F0F0Fu;   // bytes: n1 n3 n5 n7
  U4 r;
  r.a = __builtin_amdgcn_perm(K64, e, 0x04010400u);  // [n0,64,n2,64]
  r.b = __builtin_amdgcn_perm(K64, e, 0x04030402u);  // [n4,64,n6,64]
  r.c = __builtin_amdgcn_perm(K64, o, 0x04010400u);  // [n1,64,n3,64]
  r.d = __builtin_amdgcn_perm(K64, o, 0x04030402u);  // [n5,64,n7,64]
  return r;
}

// ---- Main GEMM: one wave owns a 16-wide N tile over one K segment.
// S = number of K segments (K-split for parallelism / latency hiding).
// S==1 writes the final result; S>1 writes raw f32 partials to `outp`.
template <int S>
__global__ __launch_bounds__(256) void q4l_gemm(
    const uint32_t* __restrict__ q, const _Float16* __restrict__ aws,
    const float* __restrict__ sums, const float* __restrict__ scales,
    const float* __restrict__ zeros, const float* __restrict__ bias,
    float* __restrict__ outp) {
  constexpr int CPS = KCH / S;   // chunks per segment
  constexpr int NG  = CPS / 4;   // pipeline groups of 4 chunks
  int wave = threadIdx.x >> 5;
  int L    = threadIdx.x & 31;
  int tile = blockIdx.x * 8 + wave;
  int seg  = blockIdx.y;
  int o0   = tile << 4;
  int hw   = L >> 4;             // half-wave: B lanes 0-15 -> K 0..15, 16-31 -> K 16..31
  int n    = o0 + (L & 15);      // B column for this lane
  int cstart = seg * CPS;

  const uint32_t* qb = q + (size_t)(cstart * 4 + 2 * hw) * OUT_F + n;
  const v16h*     ab = (const v16h*)aws + (size_t)cstart * 32 + L;

  uint32_t bw[8];
  v16h     af[4];
#pragma unroll
  for (int g = 0; g < 4; ++g) {
    bw[2 * g]     = qb[(size_t)(4 * g) * OUT_F];
    bw[2 * g + 1] = qb[(size_t)(4 * g + 1) * OUT_F];
    af[g]         = ab[g * 32];
  }

  v8f acc = {};
  for (int grp = 0; grp < NG - 1; ++grp) {
    const uint32_t* qn = qb + (size_t)16 * OUT_F;
    const v16h*     an = ab + 128;
    uint32_t nbw[8];
    v16h     naf[4];
#pragma unroll
    for (int g = 0; g < 4; ++g) {            // prefetch next group into regs
      nbw[2 * g]     = qn[(size_t)(4 * g) * OUT_F];
      nbw[2 * g + 1] = qn[(size_t)(4 * g + 1) * OUT_F];
      naf[g]         = an[g * 32];
    }
    if (grp + 4 < NG) {                      // HBM->L2 prefetch, 4 groups ahead
      const uint32_t* qp = qb + (size_t)64 * OUT_F;
#pragma unroll
      for (int g = 0; g < 4; ++g) {
        __builtin_prefetch(qp + (size_t)(4 * g) * OUT_F, 0, 0);
        __builtin_prefetch(qp + (size_t)(4 * g + 1) * OUT_F, 0, 0);
      }
    }
#pragma unroll
    for (int g = 0; g < 4; ++g) {            // compute current group
      U4 lo = q4l_unpack(bw[2 * g]);
      U4 hi = q4l_unpack(bw[2 * g + 1]);
      v8u bu = {lo.a, lo.b, lo.c, lo.d, hi.a, hi.b, hi.c, hi.d};
      v16h b = __builtin_bit_cast(v16h, bu);
      acc = __builtin_amdgcn_wmma_f32_16x16x32_f16(false, af[g], false, b,
                                                   (short)0, acc, false, false);
    }
#pragma unroll
    for (int i = 0; i < 8; ++i) bw[i] = nbw[i];
#pragma unroll
    for (int g = 0; g < 4; ++g) af[g] = naf[g];
    qb = qn; ab = an;
  }
#pragma unroll
  for (int g = 0; g < 4; ++g) {              // tail group
    U4 lo = q4l_unpack(bw[2 * g]);
    U4 hi = q4l_unpack(bw[2 * g + 1]);
    v8u bu = {lo.a, lo.b, lo.c, lo.d, hi.a, hi.b, hi.c, hi.d};
    v16h b = __builtin_bit_cast(v16h, bu);
    acc = __builtin_amdgcn_wmma_f32_16x16x32_f16(false, af[g], false, b,
                                                 (short)0, acc, false, false);
  }

  int mb = hw * 8;   // C/D layout: lanes 0-15 -> M=r, lanes 16-31 -> M=r+8
  if (S == 1) {
    float s = scales[n], z = zeros[n], bb = bias[n];
    float coef = 1024.0f * s + z;
#pragma unroll
    for (int r = 0; r < 8; ++r)
      outp[(size_t)(mb + r) * OUT_F + n] = acc[r] * s - sums[mb + r] * coef + bb;
  } else {
    float* part = outp + (size_t)seg * MN;
#pragma unroll
    for (int r = 0; r < 8; ++r)
      part[(size_t)(mb + r) * OUT_F + n] = acc[r];
  }
}

// ---- Deterministic cross-segment reduction + epilogue (S>1 path).
template <int S>
__global__ __launch_bounds__(256) void q4l_fixup(
    const float* __restrict__ part, const float* __restrict__ sums,
    const float* __restrict__ scales, const float* __restrict__ zeros,
    const float* __restrict__ bias, float* __restrict__ out) {
  int n = blockIdx.x * 256 + threadIdx.x;
  int m = blockIdx.y;
  size_t idx = (size_t)m * OUT_F + n;
  float a = 0.f;
#pragma unroll
  for (int s = 0; s < S; ++s) a += part[(size_t)s * MN + idx];
  float sc = scales[n];
  out[idx] = a * sc - sums[m] * (1024.0f * sc + zeros[n]) + bias[n];
}

extern "C" void kernel_launch(void* const* d_in, const int* in_sizes, int n_in,
                              void* d_out, int out_size, void* d_ws, size_t ws_size,
                              hipStream_t stream) {
  const float*    x      = (const float*)d_in[0];
  const uint32_t* qw     = (const uint32_t*)d_in[1];
  const float*    scales = (const float*)d_in[2];
  const float*    zeros  = (const float*)d_in[3];
  const float*    bias   = (const float*)d_in[4];
  float*          out    = (float*)d_out;

  const size_t A_BYTES   = 256 * 1024;          // 256 chunks * 32 lanes * 32B
  const size_t SUM_OFF   = A_BYTES;             // 16 floats
  const size_t PART_OFF  = A_BYTES + 64;
  const size_t SEG_BYTES = (size_t)MN * sizeof(float);   // 1.75 MB per segment

  _Float16* aws  = (_Float16*)d_ws;
  float*    sums = (float*)((char*)d_ws + SUM_OFF);
  float*    part = (float*)((char*)d_ws + PART_OFF);

  q4l_packA<<<32, 256, 0, stream>>>(x, aws);
  q4l_rowsum<<<1, 512, 0, stream>>>(x, sums);

  dim3 fgrid(OUT_F / 256, TOKENS);
  if (ws_size >= PART_OFF + 4 * SEG_BYTES) {
    q4l_gemm<4><<<dim3(OUT_F / 128, 4), 256, 0, stream>>>(qw, aws, sums, scales, zeros, bias, part);
    q4l_fixup<4><<<fgrid, 256, 0, stream>>>(part, sums, scales, zeros, bias, out);
  } else if (ws_size >= PART_OFF + 2 * SEG_BYTES) {
    q4l_gemm<2><<<dim3(OUT_F / 128, 2), 256, 0, stream>>>(qw, aws, sums, scales, zeros, bias, part);
    q4l_fixup<2><<<fgrid, 256, 0, stream>>>(part, sums, scales, zeros, bias, out);
  } else {
    q4l_gemm<1><<<dim3(OUT_F / 128, 1), 256, 0, stream>>>(qw, aws, sums, scales, zeros, bias, out);
  }
}
